// MultiBoxConvBlock_47708496724920
// MI455X (gfx1250) — matrix operand order repaired and checked by reading, not compiled, not run on hardware
//
#include <hip/hip_runtime.h>
#include <hip/hip_bf16.h>

// ---- problem constants ----
#define BB 4
#define CC 64
#define KK 4
#define HH 192
#define WW 192
#define HW (HH * WW)            // 36864
#define NPIX (BB * HW)          // 147456
#define IID 193                 // integral image dim (H+1)
#define IISZ (IID * IID)        // 37249 floats per plane
#define CKTOT (CC * KK)         // 256
#define WSTR 264                // padded LDS weight row stride (bf16 units)

typedef __attribute__((ext_vector_type(16))) __bf16 v16bf;
typedef __attribute__((ext_vector_type(8)))  float  v8f;

union Frag16 { v16bf v; float4 q[2]; };

__device__ __forceinline__ __bf16 f2bf(float f) { return (__bf16)f; }

// ---------------------------------------------------------------------------
// Kernel 1: convert 1x1-conv weights (64 x 256 f32, row-major [o][ck]) to bf16
// ---------------------------------------------------------------------------
__global__ void pack_weights_kernel(const float* __restrict__ w1,
                                    const float* __restrict__ w2,
                                    __bf16* __restrict__ p1,
                                    __bf16* __restrict__ p2) {
    int i = blockIdx.x * blockDim.x + threadIdx.x;
    if (i < CC * CKTOT) {
        p1[i] = f2bf(w1[i]);
        p2[i] = f2bf(w2[i]);
    }
}

// ---------------------------------------------------------------------------
// Kernel 2: relu + integral image, one (b,c) plane per block.
// II plane is (193 x 193), row/col 0 are zero, II[i][j] = sum x[0..i-1][0..j-1].
// Pass 1: per-row prefix (thread = row). Pass 2: per-col prefix (coalesced).
// relu is idempotent, so reusing this for already-relu'd h1 is exact.
// ---------------------------------------------------------------------------
__global__ __launch_bounds__(256)
void integral_kernel(const float* __restrict__ src, float* __restrict__ II) {
    const int plane = blockIdx.x;                  // 0 .. B*C-1
    const float* s = src + (size_t)plane * HW;
    float* p = II + (size_t)plane * IISZ;
    const int t = threadIdx.x;

    if (t < IID) p[t] = 0.0f;                      // row 0 = 0
    if (t < HH) {
        const float* r = s + (size_t)t * WW;
        float* o = p + (size_t)(t + 1) * IID;
        o[0] = 0.0f;                               // col 0 = 0
        float acc = 0.0f;
        for (int j = 0; j < WW; ++j) {
            acc += fmaxf(r[j], 0.0f);
            o[j + 1] = acc;
        }
    }
    __threadfence();
    __syncthreads();
    if (t < IID) {
        float* col = p + t;
        float acc = 0.0f;
        for (int i = 1; i < IID; ++i) {
            acc += col[(size_t)i * IID];
            col[(size_t)i * IID] = acc;
        }
    }
}

// ---------------------------------------------------------------------------
// Kernel 3: fractional box filter from integral images.
// Block = 256 threads covering 64 consecutive x pixels (tx) x 4 ck slots (tg),
// looping ck += 4 so each block produces a full 64px x 256ck tile.
// Lanes of a wave share ck and span consecutive x -> coalesced II taps.
// Results staged in LDS (bf16), then written coalesced as [pix][ck] rows,
// exactly the A-operand layout the WMMA GEMM consumes.
// ---------------------------------------------------------------------------
__global__ __launch_bounds__(256)
void boxeval_kernel(const float* __restrict__ II,
                    const float* __restrict__ ymin, const float* __restrict__ ymax,
                    const float* __restrict__ xmin, const float* __restrict__ xmax,
                    __bf16* __restrict__ hbox) {
    __shared__ __bf16 tile[64][CKTOT];             // 32 KB

    const int tx = threadIdx.x & 63;               // x within tile
    const int tg = threadIdx.x >> 6;               // 0..3
    const int xTile = blockIdx.x;                  // 0..2
    const int by = blockIdx.y;                     // b*H + y
    const int b = by / HH;
    const int y = by - b * HH;
    const int xg = xTile * 64 + tx;
    const int pixBase = (b * HH + y) * WW + xTile * 64;

    const float fy_ = (float)y + 0.5f;
    const float fx_ = (float)xg + 0.5f;

    for (int ck = tg; ck < CKTOT; ck += 4) {
        const int c = ck >> 2;
        const float* P = II + (size_t)(b * CC + c) * IISZ;

        const float ymn = ymin[ck], ymx = ymax[ck];
        const float xmn = xmin[ck], xmx = xmax[ck];
        const float invArea = 1.0f / ((ymx - ymn) * (xmx - xmn));

        // y (row) interpolation coordinates -- uniform across the wave
        float yct = fminf(fmaxf(fy_ + ymn, 0.0f), 192.0f);
        float y0tf = fminf(floorf(yct), 191.0f);
        float fyt = yct - y0tf;
        int y0t = (int)y0tf;

        float ycb = fminf(fmaxf(fy_ + ymx, 0.0f), 192.0f);
        float y0bf = fminf(floorf(ycb), 191.0f);
        float fyb = ycb - y0bf;
        int y0b = (int)y0bf;

        // x (col) interpolation coordinates -- per lane
        float xcl = fminf(fmaxf(fx_ + xmn, 0.0f), 192.0f);
        float x0lf = fminf(floorf(xcl), 191.0f);
        float fxl = xcl - x0lf;
        int x0l = (int)x0lf;

        float xcr = fminf(fmaxf(fx_ + xmx, 0.0f), 192.0f);
        float x0rf = fminf(floorf(xcr), 191.0f);
        float fxr = xcr - x0rf;
        int x0r = (int)x0rf;

        const float* rt0 = P + (size_t)y0t * IID;
        const float* rt1 = rt0 + IID;
        const float* rb0 = P + (size_t)y0b * IID;
        const float* rb1 = rb0 + IID;

        auto dcol = [&](int j) -> float {
            return (1.0f - fyb) * rb0[j] + fyb * rb1[j]
                 - (1.0f - fyt) * rt0[j] - fyt * rt1[j];
        };

        float left  = (1.0f - fxl) * dcol(x0l) + fxl * dcol(x0l + 1);
        float right = (1.0f - fxr) * dcol(x0r) + fxr * dcol(x0r + 1);

        tile[tx][ck] = f2bf((right - left) * invArea);
    }
    __syncthreads();

    // coalesced write-out: 64 rows x 512 bytes, as u32 lanes
    const unsigned* t32 = (const unsigned*)&tile[0][0];
    unsigned* dst = (unsigned*)hbox;
    for (int i = threadIdx.x; i < 64 * (CKTOT / 2); i += 256) {
        int p = i >> 7;                            // pixel within tile
        int col = i & 127;                         // u32 column
        dst[(size_t)(pixBase + p) * (CKTOT / 2) + col] = t32[i];
    }
}

// ---------------------------------------------------------------------------
// Kernel 4: 1x1 conv as WMMA GEMM.  out[pix][o] = relu( A[pix][:] . Wp[o][:]
//           + bias[o] (+ residual) ), written to NCHW plane layout.
// Weights are staged once per block in LDS (rows padded to 264 bf16 so the
// per-lane ds_load_b128 B-fragment reads rotate across all 64 banks).
// Each wave computes 2 M-tiles (32 pixels) x 64 channels: every B fragment
// feeds 2 WMMAs -> 64 v_wmma_f32_16x16x32_bf16 per wave.
// Fragment layouts per CDNA5 ISA 7.12.2 (wave32):
//   A 16x32 bf16 : lane = M | (khalf<<4); elems 0..7 = K[kh..kh+7],
//                  elems 8..15 = K[16+kh..16+kh+7], kh = 8*(lane>=16)
//   B 32x16 bf16 : lane = N | (khalf<<4); same K order as A, from weight row N
//   C/D 16x16 f32: vgpr r -> M = r + 8*(lane>=16), N = lane & 15
// ---------------------------------------------------------------------------
__global__ __launch_bounds__(256)
void conv1x1_wmma_kernel(const __bf16* __restrict__ A,
                         const __bf16* __restrict__ Wp,
                         const float* __restrict__ bias,
                         const float* __restrict__ residual,   // NCHW or null
                         float* __restrict__ out,              // NCHW
                         int hasRes) {
    __shared__ __bf16 wlds[CC * WSTR];             // 64 rows x 264 bf16 = 33 KB

    // cooperative stage of the 64x256 weight matrix into padded LDS rows
    {
        const float4* src = (const float4*)Wp;     // 2048 float4
        for (int i = threadIdx.x; i < CC * (CKTOT / 8); i += 256) {
            int row = i >> 5;                      // / 32 float4 per row
            int col = i & 31;
            *((float4*)(wlds + row * WSTR) + col) = src[i];
        }
    }
    __syncthreads();

    const int wave = blockIdx.x * 8 + (threadIdx.x >> 5);
    const int lane = threadIdx.x & 31;
    const int n0 = lane & 15;
    const int hi = lane >> 4;                      // K-half selector

    const __bf16* Arow0 = A + (size_t)(wave * 32 + n0) * CKTOT + hi * 8;  // M-tile 0
    const __bf16* Arow1 = Arow0 + (size_t)16 * CKTOT;                     // M-tile 1
    const __bf16* Blds  = wlds + n0 * WSTR + hi * 8;

    v8f acc[8] = {};                               // [mt*4 + ntile]

#pragma unroll
    for (int kk = 0; kk < CKTOT; kk += 32) {
        Frag16 a0, a1;
        a0.q[0] = *reinterpret_cast<const float4*>(Arow0 + kk);
        a0.q[1] = *reinterpret_cast<const float4*>(Arow0 + kk + 16);
        a1.q[0] = *reinterpret_cast<const float4*>(Arow1 + kk);
        a1.q[1] = *reinterpret_cast<const float4*>(Arow1 + kk + 16);

#pragma unroll
        for (int t = 0; t < 4; ++t) {
            Frag16 bf;
            const __bf16* bp = Blds + t * 16 * WSTR + kk;
            bf.q[0] = *reinterpret_cast<const float4*>(bp);
            bf.q[1] = *reinterpret_cast<const float4*>(bp + 16);
            acc[t] = __builtin_amdgcn_wmma_f32_16x16x32_bf16(
                false, a0.v, false, bf.v, (short)0, acc[t], false, false);
            acc[4 + t] = __builtin_amdgcn_wmma_f32_16x16x32_bf16(
                false, a1.v, false, bf.v, (short)0, acc[4 + t], false, false);
        }
    }

#pragma unroll
    for (int mt = 0; mt < 2; ++mt) {
        const int mbase = wave * 32 + mt * 16 + hi * 8;
#pragma unroll
        for (int r = 0; r < 8; ++r) {
            const int pix = mbase + r;
            const int bIdx = pix / HW;
            const int rem = pix - bIdx * HW;
#pragma unroll
            for (int t = 0; t < 4; ++t) {
                const int o = t * 16 + n0;
                float v = acc[mt * 4 + t][r] + bias[o];
                const size_t addr = (size_t)(bIdx * CC + o) * HW + rem;
                if (hasRes) v += residual[addr];
                out[addr] = fmaxf(v, 0.0f);
            }
        }
    }
}

// ---------------------------------------------------------------------------
extern "C" void kernel_launch(void* const* d_in, const int* in_sizes, int n_in,
                              void* d_out, int out_size, void* d_ws, size_t ws_size,
                              hipStream_t stream) {
    const float* x     = (const float*)d_in[0];
    const float* ymin1 = (const float*)d_in[1];
    const float* ymax1 = (const float*)d_in[2];
    const float* xmin1 = (const float*)d_in[3];
    const float* xmax1 = (const float*)d_in[4];
    const float* w1    = (const float*)d_in[5];
    const float* b1    = (const float*)d_in[6];
    const float* ymin2 = (const float*)d_in[7];
    const float* ymax2 = (const float*)d_in[8];
    const float* xmin2 = (const float*)d_in[9];
    const float* xmax2 = (const float*)d_in[10];
    const float* w2    = (const float*)d_in[11];
    const float* b2    = (const float*)d_in[12];
    float* out = (float*)d_out;
    (void)in_sizes; (void)n_in; (void)out_size; (void)ws_size;

    // workspace carve-out (256B aligned)
    char* ws = (char*)d_ws;
    size_t off = 0;
    auto take = [&](size_t bytes) -> void* {
        void* p = ws + off;
        off = (off + bytes + 255) & ~(size_t)255;
        return p;
    };
    float*  II   = (float*) take((size_t)BB * CC * IISZ * sizeof(float));   // ~38 MB
    __bf16* hbox = (__bf16*)take((size_t)NPIX * CKTOT * sizeof(__bf16));    // ~75 MB
    float*  h1   = (float*) take((size_t)NPIX * CC * sizeof(float));        // ~38 MB
    __bf16* wp1  = (__bf16*)take((size_t)CC * CKTOT * sizeof(__bf16));
    __bf16* wp2  = (__bf16*)take((size_t)CC * CKTOT * sizeof(__bf16));

    const int gemmBlocks = NPIX / 32 / 8;          // 32 rows/wave, 8 waves/block

    pack_weights_kernel<<<(CC * CKTOT + 255) / 256, 256, 0, stream>>>(w1, w2, wp1, wp2);

    // stage 1: relu(x) -> II -> box -> conv1(+relu) -> h1 (NCHW)
    integral_kernel<<<BB * CC, 256, 0, stream>>>(x, II);
    boxeval_kernel<<<dim3(WW / 64, BB * HH), 256, 0, stream>>>(II, ymin1, ymax1,
                                                               xmin1, xmax1, hbox);
    conv1x1_wmma_kernel<<<gemmBlocks, 256, 0, stream>>>(hbox, wp1, b1, nullptr, h1, 0);

    // stage 2: h1 -> II -> box -> conv2 + residual + relu -> out (NCHW)
    integral_kernel<<<BB * CC, 256, 0, stream>>>(h1, II);
    boxeval_kernel<<<dim3(WW / 64, BB * HH), 256, 0, stream>>>(II, ymin2, ymax2,
                                                               xmin2, xmax2, hbox);
    conv1x1_wmma_kernel<<<gemmBlocks, 256, 0, stream>>>(hbox, wp2, b2, x, out, 1);
}
